// ModelEdgeGIN_13932873908324
// MI455X (gfx1250) — matrix operand order repaired and checked by reading, not compiled, not run on hardware
//
#include <hip/hip_runtime.h>
#include <hip/hip_bf16.h>

typedef __attribute__((ext_vector_type(16))) __bf16 v16bf;
typedef __attribute__((ext_vector_type(8)))  float  v8f;

#define HID        128
#define NNODES     100000
#define NEDGES     1600000
#define NGRAPHS    512
#define BN_EPS_C   1e-5f

static __device__ __forceinline__ void atomic_add_f32(float* p, float v) {
    __hip_atomic_fetch_add(p, v, __ATOMIC_RELAXED, __HIP_MEMORY_SCOPE_AGENT);
}

// ---------------------------------------------------------------------------
// Pack a row-major f32 weight [K x N] into bf16 WMMA B-fragments.
// Fragment (kstep, ntile) holds B[kstep*32 .. +32][ntile*16 .. +16].
// Per ISA 16-bit B layout (wave32): col = lane & 15; klo = 0 for lanes 0-15,
// 8 for lanes 16-31; v16bf element e<8 -> K = klo + e, e>=8 -> K = 16 + klo + (e-8).
// Each lane's 16 bf16 (32 B) stored contiguously -> b128-friendly loads.
// K beyond the real extent is zero-padded (used for the K=16 edge linear).
// ---------------------------------------------------------------------------
__global__ void pack_weight_kernel(const float* __restrict__ W,
                                   __bf16* __restrict__ pack,
                                   int K, int N, int total) {
    int idx = blockIdx.x * blockDim.x + threadIdx.x;
    if (idx >= total) return;
    int e      = idx & 15;
    int lane   = (idx >> 4) & 31;
    int frag   = idx >> 9;
    int ntiles = N >> 4;
    int nt     = frag % ntiles;
    int kstep  = frag / ntiles;
    unsigned n = nt * 16 + (lane & 15);
    int k = kstep * 32 + ((lane >= 16) ? 8 : 0) + (e & 7) + ((e >= 8) ? 16 : 0);
    float v = (k < K) ? W[(unsigned)k * (unsigned)N + n] : 0.0f;
    pack[idx] = (__bf16)v;
}

__global__ void fill0_kernel(float* __restrict__ p, unsigned n) {
    unsigned i = blockIdx.x * blockDim.x + threadIdx.x;
    unsigned stride = gridDim.x * blockDim.x;
    for (; i < n; i += stride) p[i] = 0.0f;
}

// ---------------------------------------------------------------------------
// Fused GINE edge phase: per wave, 16 edges.
//   e  = edge_attr[16x16] @ We[16x128] + be      (WMMA, K zero-padded to 32)
//   m  = relu(x[src] + e)                        (gathered add in epilogue)
//   agg[dst] += m                                (global_atomic_add_f32)
// All global indices are 32-bit element offsets so the backend can use
// saddr + scaled-voffset addressing (no 64-bit VALU address chains).
// ---------------------------------------------------------------------------
__global__ void __launch_bounds__(128)
edge_kernel(const float* __restrict__ h,
            const float* __restrict__ ea,
            const int*   __restrict__ src,
            const int*   __restrict__ dst,
            const __bf16* __restrict__ packWe,
            const float* __restrict__ be,
            float* __restrict__ agg) {
    const int lane  = threadIdx.x & 31;
    const int wave  = threadIdx.x >> 5;
    const unsigned ebase = (blockIdx.x * 4 + wave) * 16;   // grid sized exactly

    // A fragment: 16 edges x 32 (real K=16). Per ISA 16-bit A layout:
    // element e<8 -> K = klo + e (klo = 0 for lanes 0-15, 8 for 16-31);
    // element e>=8 -> K >= 16 -> zero pad.
    const unsigned arow = lane & 15;
    const unsigned koff = (lane >= 16) ? 8u : 0u;
    const unsigned eaOff = (ebase + arow) * 16u + koff;
    const float4 f0 = *(const float4*)(ea + eaOff);
    const float4 f1 = *(const float4*)(ea + eaOff + 4u);
    v16bf a;
    a[0] = (__bf16)f0.x; a[1] = (__bf16)f0.y; a[2] = (__bf16)f0.z; a[3] = (__bf16)f0.w;
    a[4] = (__bf16)f1.x; a[5] = (__bf16)f1.y; a[6] = (__bf16)f1.z; a[7] = (__bf16)f1.w;
#pragma unroll
    for (int e = 8; e < 16; ++e) a[e] = (__bf16)0.0f;

    // Per-lane edge indices for the 8 rows this lane touches in D tiles,
    // pre-scaled to 32-bit element offsets (<< 7 = *HID).
    const unsigned mrow = (lane >= 16) ? 8u : 0u;
    unsigned sOff[8], dOff[8];
#pragma unroll
    for (int r = 0; r < 8; ++r) {
        sOff[r] = ((unsigned)src[ebase + mrow + r]) << 7;
        dOff[r] = ((unsigned)dst[ebase + mrow + r]) << 7;
    }

    const unsigned c0 = lane & 15;
    const v8f zero8 = {0.f,0.f,0.f,0.f,0.f,0.f,0.f,0.f};
#pragma unroll
    for (int nt = 0; nt < 8; ++nt) {
        const v16bf b = *(const v16bf*)(packWe + (unsigned)(nt * 32 + lane) * 16u);
        v8f d = __builtin_amdgcn_wmma_f32_16x16x32_bf16(
            false, a, false, b, (short)0, zero8, false, false);
        const unsigned c = nt * 16 + c0;
        const float bia = be[c];
#pragma unroll
        for (int r = 0; r < 8; ++r) {
            float xv = h[sOff[r] + c];
            float v  = d[r] + bia + xv;
            v = fmaxf(v, 0.0f);
            atomic_add_f32(agg + (dOff[r] + c), v);
        }
    }
}

// ---------------------------------------------------------------------------
// 128-wide GEMM: out = epilogue((A0 [+ A1]) @ Wpack + bias), N = K = 128.
// One wave computes a 16x128 strip: 8 accumulator tiles, 4 K-steps,
// 32 v_wmma_f32_16x16x32_bf16 per wave. The 32 KB weight pack is staged
// once per block into LDS (4 waves share it -> 4x less L2 traffic);
// fragments then come from ds_load_b128. Tail waves clamp their row base
// (duplicate identical stores) so EXEC stays all-ones and __syncthreads is
// uniform. Epilogue: bias (+BN) + ReLU.
// ---------------------------------------------------------------------------
template <bool HAS_ADD, bool HAS_BN>
__global__ void __launch_bounds__(128)
gemm128_kernel(const float* __restrict__ A0,
               const float* __restrict__ A1,
               const __bf16* __restrict__ pack,
               const float* __restrict__ bias,
               const float* __restrict__ gamma,
               const float* __restrict__ beta,
               const float* __restrict__ rm,
               const float* __restrict__ rv,
               float* __restrict__ out, int M) {
    __shared__ __align__(64) uint4 sB[2048];        // 32 KB: full B pack in LDS
    {
        const uint4* gp = (const uint4*)pack;
#pragma unroll
        for (int t = 0; t < 16; ++t)
            sB[threadIdx.x + t * 128] = gp[threadIdx.x + t * 128];
    }
    __syncthreads();
    const __bf16* sBp = (const __bf16*)sB;

    const int lane = threadIdx.x & 31;
    const int wave = threadIdx.x >> 5;
    int rowBase = (blockIdx.x * 4 + wave) * 16;
    if (rowBase > M - 16) rowBase = M - 16;         // clamp: wave stays fully active

    const v8f zero8 = {0.f,0.f,0.f,0.f,0.f,0.f,0.f,0.f};
    v8f acc[8];
#pragma unroll
    for (int i = 0; i < 8; ++i) acc[i] = zero8;

    const unsigned arow = (unsigned)rowBase + (lane & 15);
    const unsigned klo8 = (lane >= 16) ? 8u : 0u;
    const unsigned rptr = arow * (unsigned)HID;

#pragma unroll
    for (int ks = 0; ks < 4; ++ks) {
        const unsigned kb = rptr + ks * 32 + klo8;
        float4 f0 = *(const float4*)(A0 + kb);
        float4 f1 = *(const float4*)(A0 + kb + 4u);
        float4 f2 = *(const float4*)(A0 + kb + 16u);
        float4 f3 = *(const float4*)(A0 + kb + 20u);
        if (HAS_ADD) {
            const float4 g0 = *(const float4*)(A1 + kb);
            const float4 g1 = *(const float4*)(A1 + kb + 4u);
            const float4 g2 = *(const float4*)(A1 + kb + 16u);
            const float4 g3 = *(const float4*)(A1 + kb + 20u);
            f0.x += g0.x; f0.y += g0.y; f0.z += g0.z; f0.w += g0.w;
            f1.x += g1.x; f1.y += g1.y; f1.z += g1.z; f1.w += g1.w;
            f2.x += g2.x; f2.y += g2.y; f2.z += g2.z; f2.w += g2.w;
            f3.x += g3.x; f3.y += g3.y; f3.z += g3.z; f3.w += g3.w;
        }
        v16bf a;
        a[0]  = (__bf16)f0.x; a[1]  = (__bf16)f0.y; a[2]  = (__bf16)f0.z; a[3]  = (__bf16)f0.w;
        a[4]  = (__bf16)f1.x; a[5]  = (__bf16)f1.y; a[6]  = (__bf16)f1.z; a[7]  = (__bf16)f1.w;
        a[8]  = (__bf16)f2.x; a[9]  = (__bf16)f2.y; a[10] = (__bf16)f2.z; a[11] = (__bf16)f2.w;
        a[12] = (__bf16)f3.x; a[13] = (__bf16)f3.y; a[14] = (__bf16)f3.z; a[15] = (__bf16)f3.w;

#pragma unroll
        for (int nt = 0; nt < 8; ++nt) {
            const v16bf b = *(const v16bf*)(sBp + (unsigned)((ks * 8 + nt) * 32 + lane) * 16u);
            acc[nt] = __builtin_amdgcn_wmma_f32_16x16x32_bf16(
                false, a, false, b, (short)0, acc[nt], false, false);
        }
    }

    const unsigned c0   = lane & 15;
    const unsigned rOff = (lane >= 16) ? 8u : 0u;
    const unsigned obase = ((unsigned)rowBase + rOff) * (unsigned)HID;
#pragma unroll
    for (int nt = 0; nt < 8; ++nt) {
        const unsigned c = nt * 16 + c0;
        const float bia = bias[c];
        float scale = 1.0f, shift = 0.0f;
        if (HAS_BN) {
            const float inv = rsqrtf(rv[c] + BN_EPS_C);
            scale = inv * gamma[c];
            shift = beta[c] - rm[c] * scale;
        }
#pragma unroll
        for (int r = 0; r < 8; ++r) {
            float v = acc[nt][r] + bia;
            if (HAS_BN) v = v * scale + shift;
            v = fmaxf(v, 0.0f);
            out[obase + (unsigned)r * (unsigned)HID + c] = v;
        }
    }
}

// global_add_pool: g[batch[v]] += h[v] ; one thread handles 4 channels.
__global__ void pool_kernel(const float* __restrict__ h,
                            const int*   __restrict__ batch,
                            float* __restrict__ g) {
    unsigned idx = blockIdx.x * blockDim.x + threadIdx.x;
    if (idx >= NNODES * 32u) return;
    const unsigned v = idx >> 5;
    const unsigned c = (idx & 31) * 4;
    const float4 hv = *(const float4*)(h + v * (unsigned)HID + c);
    const unsigned b = (unsigned)batch[v];
    float* gp = g + b * (unsigned)HID + c;
    atomic_add_f32(gp + 0, hv.x);
    atomic_add_f32(gp + 1, hv.y);
    atomic_add_f32(gp + 2, hv.z);
    atomic_add_f32(gp + 3, hv.w);
}

// out = t2 @ Wo2[:,0] + bo2   (512 x 128 -> 512)
__global__ void out2_kernel(const float* __restrict__ t2,
                            const float* __restrict__ W2,
                            const float* __restrict__ b2,
                            float* __restrict__ out) {
    unsigned g = blockIdx.x * blockDim.x + threadIdx.x;
    if (g >= NGRAPHS) return;
    float s = b2[0];
#pragma unroll 8
    for (unsigned c = 0; c < HID; ++c) s += t2[g * (unsigned)HID + c] * W2[c];
    out[g] = s;
}

extern "C" void kernel_launch(void* const* d_in, const int* in_sizes, int n_in,
                              void* d_out, int out_size, void* d_ws, size_t ws_size,
                              hipStream_t stream) {
    (void)n_in; (void)out_size; (void)ws_size;
    auto F = [&](int i) { return (const float*)d_in[i]; };
    auto I = [&](int i) { return (const int*)d_in[i]; };

    const float* x;  const float* ea;
    const int *src, *dst, *batch;
    struct Layer { const float *We,*be,*W1,*b1,*W2,*b2,*gamma,*beta,*rm,*rv; } L[3];
    const float *Wo1, *bo1, *Wo2, *bo2;

    if (in_sizes[0] == NNODES * HID) {
        // Insertion-order flatten: x, edge_index, edge_attr, batch,
        // c1{We,be,W1,b1,W2,b2}, bn1{gamma,beta,rm,rv}, c2, bn2, c3, bn3,
        // out{W1,b1,W2,b2}
        x = F(0);
        const int* ei = I(1);
        ea = F(2); batch = I(3);
        src = ei; dst = ei + NEDGES;
        for (int i = 0; i < 3; ++i) {
            const int b = 4 + 10 * i;
            L[i].We = F(b+0); L[i].be = F(b+1);
            L[i].W1 = F(b+2); L[i].b1 = F(b+3);
            L[i].W2 = F(b+4); L[i].b2 = F(b+5);
            L[i].gamma = F(b+6); L[i].beta = F(b+7);
            L[i].rm = F(b+8);    L[i].rv = F(b+9);
        }
        Wo1 = F(34); bo1 = F(35); Wo2 = F(36); bo2 = F(37);
    } else {
        // JAX sorted tree-flatten: batch, edge_attr, edge_index,
        // bn1{beta,gamma,rm,rv}, bn2, bn3, c1{W1,W2,We,b1,b2,be}, c2, c3,
        // out{W1,W2,b1,b2}, x
        batch = I(0); ea = F(1);
        const int* ei = I(2);
        src = ei; dst = ei + NEDGES;
        for (int i = 0; i < 3; ++i) {
            const int bb = 3 + 4 * i;
            const int cb = 15 + 6 * i;
            L[i].beta = F(bb+0); L[i].gamma = F(bb+1);
            L[i].rm   = F(bb+2); L[i].rv    = F(bb+3);
            L[i].W1 = F(cb+0); L[i].W2 = F(cb+1); L[i].We = F(cb+2);
            L[i].b1 = F(cb+3); L[i].b2 = F(cb+4); L[i].be = F(cb+5);
        }
        Wo1 = F(33); Wo2 = F(34); bo1 = F(35); bo2 = F(36);
        x = F(37);
    }

    // Workspace carve (256-B aligned)
    char* ws = (char*)d_ws;
    size_t off = 0;
    auto carve = [&](size_t bytes) {
        char* p = ws + off;
        off = (off + bytes + 255) & ~(size_t)255;
        return (void*)p;
    };
    float* agg  = (float*)carve((size_t)NNODES * HID * 4);
    float* tbuf = (float*)carve((size_t)NNODES * HID * 4);
    float* hbuf = (float*)carve((size_t)NNODES * HID * 4);
    float* gbuf = (float*)carve((size_t)NGRAPHS * HID * 4);
    float* t2   = (float*)carve((size_t)NGRAPHS * HID * 4);
    __bf16* pWe[3]; __bf16* pW1[3]; __bf16* pW2[3];
    for (int i = 0; i < 3; ++i) {
        pWe[i] = (__bf16*)carve((size_t)1 * 8 * 512 * 2);
        pW1[i] = (__bf16*)carve((size_t)4 * 8 * 512 * 2);
        pW2[i] = (__bf16*)carve((size_t)4 * 8 * 512 * 2);
    }
    __bf16* pWo1 = (__bf16*)carve((size_t)4 * 8 * 512 * 2);

    // Pack weights -> bf16 B-fragment layout (L2-resident, ~250 KB total)
    for (int i = 0; i < 3; ++i) {
        pack_weight_kernel<<<(4096 + 255) / 256, 256, 0, stream>>>(L[i].We, pWe[i], 16, HID, 4096);
        pack_weight_kernel<<<(16384 + 255) / 256, 256, 0, stream>>>(L[i].W1, pW1[i], HID, HID, 16384);
        pack_weight_kernel<<<(16384 + 255) / 256, 256, 0, stream>>>(L[i].W2, pW2[i], HID, HID, 16384);
    }
    pack_weight_kernel<<<(16384 + 255) / 256, 256, 0, stream>>>(Wo1, pWo1, HID, HID, 16384);

    const unsigned nAgg = (unsigned)NNODES * HID;
    const int gemmGrid = (NNODES / 16 + 3) / 4;   // 6250 waves -> 1563 blocks
    const float* hin = x;
    for (int i = 0; i < 3; ++i) {
        fill0_kernel<<<2048, 256, 0, stream>>>(agg, nAgg);
        edge_kernel<<<NEDGES / 64, 128, 0, stream>>>(hin, ea, src, dst, pWe[i], L[i].be, agg);
        gemm128_kernel<true, false><<<gemmGrid, 128, 0, stream>>>(
            hin, agg, pW1[i], L[i].b1, nullptr, nullptr, nullptr, nullptr, tbuf, NNODES);
        gemm128_kernel<false, true><<<gemmGrid, 128, 0, stream>>>(
            tbuf, nullptr, pW2[i], L[i].b2, L[i].gamma, L[i].beta, L[i].rm, L[i].rv, hbuf, NNODES);
        hin = hbuf;
    }

    fill0_kernel<<<256, 256, 0, stream>>>(gbuf, (unsigned)NGRAPHS * HID);
    pool_kernel<<<(NNODES * 32 + 255) / 256, 256, 0, stream>>>(hin, batch, gbuf);
    gemm128_kernel<false, false><<<(NGRAPHS / 16 + 3) / 4, 128, 0, stream>>>(
        gbuf, nullptr, pWo1, bo1, nullptr, nullptr, nullptr, nullptr, t2, NGRAPHS);
    out2_kernel<<<2, 256, 0, stream>>>(t2, Wo2, bo2, (float*)d_out);
}
